// SPINN_37014028157116
// MI455X (gfx1250) — matrix operand order repaired and checked by reading, not compiled, not run on hardware
//
#include <hip/hip_runtime.h>

// ---------------------------------------------------------------------------
// SPINN fused tracker step for MI455X (gfx1250, wave32, WMMA).
//   gates[4096,4096] = concat(Xb,Xs1,Xs2,Hprev) @ concat(Wb,Ws1,Ws2,Wlat) + b
//   c = tanh(a)*sig(i) + sig(f)*c_prev ; h = sig(o)*tanh(c)
// bf16 WMMA GEMM (f32 accum), double-buffered LDS, register prefetch,
// per-thread base pointer + immediate-offset addressing, fused LSTM epilogue.
// ---------------------------------------------------------------------------

typedef __attribute__((ext_vector_type(16))) __bf16 v16bf;
typedef __attribute__((ext_vector_type(8)))  __bf16 v8bf;
typedef __attribute__((ext_vector_type(4)))  __bf16 v4bf;
typedef __attribute__((ext_vector_type(8)))  float  v8f;
typedef __attribute__((ext_vector_type(4)))  float  f32x4;

#define BATCH   4096
#define KDIM    1024    // per-term K (SIZE == TRACK)
#define TRACKN  1024
#define NGATE   4096    // 4*TRACK
#define BM      128     // rows per block
#define BN      32      // gate cols per block per chunk (x4 chunks = 128)
#define BK      32      // K per step (== WMMA K)
#define LDS_LD  40      // padded row stride in bf16 elems (80 B, 16B-aligned)
#define NSTEPS  128     // 4 terms * (KDIM/BK)

__device__ __forceinline__ float fast_sigmoid(float x) {
    return __builtin_amdgcn_rcpf(1.0f + __expf(-x));
}
__device__ __forceinline__ float fast_tanh(float x) {
    // tanh(x) = 1 - 2/(exp(2x)+1)
    return 1.0f - 2.0f * __builtin_amdgcn_rcpf(__expf(2.0f * x) + 1.0f);
}

__global__ __launch_bounds__(256)
void spinn_tracker_wmma(const float* __restrict__ x0, const float* __restrict__ x1,
                        const float* __restrict__ x2, const float* __restrict__ x3,
                        const float* __restrict__ w0, const float* __restrict__ w1,
                        const float* __restrict__ w2, const float* __restrict__ w3,
                        const float* __restrict__ bias, const float* __restrict__ c_prev,
                        float* __restrict__ h_out, float* __restrict__ c_out)
{
    // A wants [M][K] (K-contiguous) -> natural for X.
    // B wants [N][K] (K-contiguous) -> W transposed during the GLOBAL load
    // (dword loads along K, coalesced across lanes along N); LDS stores are b64.
    __shared__ __align__(16) __bf16 lds_x[2][BM][LDS_LD];       // [buf][row][k]
    __shared__ __align__(16) __bf16 lds_w[2][4 * BN][LDS_LD];   // [buf][n(chunk-major)][k]

    const int tid  = threadIdx.x;
    const int lane = tid & 31;
    const int wave = tid >> 5;     // 0..7
    const int wm   = wave >> 1;    // 0..3 -> 32-row group
    const int wn   = wave & 1;     // 0..1 -> 16-column group
    const int half = lane >> 4;    // A: K-phase, B: K-group, C/D: M+8
    const int l15  = lane & 15;

    const int m0 = blockIdx.y * BM;   // batch rows
    const int n0 = blockIdx.x * BN;   // gate columns within each chunk

    // ---- fetch geometry: ONE per-thread pointer per matrix + imm offsets ---
    // X: row_i = (tid>>3)+32i (imm: 32*KDIM elems), col = (tid&7)*4
    const int xoff = (m0 + (tid >> 3)) * KDIM + (tid & 7) * 4;      // elements
    // W: n = tid&127 (i-indep), kq_i = 4*(tid>>7)+8i ((8i+e)*NGATE is imm)
    const int wnl  = tid & 127;                                     // LDS n index
    const int wkq  = 4 * (tid >> 7);                                // k base
    const int woff = wkq * NGATE + (wnl >> 5) * TRACKN + n0 + (wnl & 31);

    auto selX = [&](int t) -> const float* {
        return t == 0 ? x0 : (t == 1 ? x1 : (t == 2 ? x2 : x3));
    };
    auto selW = [&](int t) -> const float* {
        return t == 0 ? w0 : (t == 1 ? w1 : (t == 2 ? w2 : w3));
    };

    float xr[4][4];   // prefetched X values (f32)
    float wr[4][4];   // prefetched W values (f32)

    auto fetch = [&](int s) {
        const int t    = s >> 5;
        const int koff = (s & 31) * BK;
        // one 64-bit add per step each; all loads below use instruction imms
        const float* __restrict__ Xt = selX(t) + koff + (size_t)xoff;
        const float* __restrict__ Wt = selW(t) + (size_t)koff * NGATE + (size_t)woff;
#pragma unroll
        for (int i = 0; i < 4; ++i) {            // 4 x global_load_b128, imm offs
            const f32x4 v = *(const f32x4*)(Xt + i * 32 * KDIM);
            xr[i][0] = v[0]; xr[i][1] = v[1]; xr[i][2] = v[2]; xr[i][3] = v[3];
        }
#pragma unroll
        for (int i = 0; i < 4; ++i)              // 16 coalesced b32, imm offs
#pragma unroll
            for (int e = 0; e < 4; ++e)
                wr[i][e] = Wt[(8 * i + e) * NGATE];
    };

    auto stash = [&](int p) {                    // regs -> LDS (v_cvt_pk_bf16_f32)
#pragma unroll
        for (int i = 0; i < 4; ++i) {
            v4bf b;
            b[0] = (__bf16)xr[i][0]; b[1] = (__bf16)xr[i][1];
            b[2] = (__bf16)xr[i][2]; b[3] = (__bf16)xr[i][3];
            *(v4bf*)&lds_x[p][(tid >> 3) + 32 * i][(tid & 7) * 4] = b;  // ds_store_b64
        }
#pragma unroll
        for (int i = 0; i < 4; ++i) {
            v4bf b;
            b[0] = (__bf16)wr[i][0]; b[1] = (__bf16)wr[i][1];
            b[2] = (__bf16)wr[i][2]; b[3] = (__bf16)wr[i][3];
            *(v4bf*)&lds_w[p][wnl][wkq + 8 * i] = b;                    // ds_store_b64
        }
    };

    v8f acc[2][4];                    // [m-tile][gate-chunk]
    const v8f vzero = {0.f, 0.f, 0.f, 0.f, 0.f, 0.f, 0.f, 0.f};
#pragma unroll
    for (int mt = 0; mt < 2; ++mt)
#pragma unroll
        for (int g = 0; g < 4; ++g) acc[mt][g] = vzero;

    // ---- software pipeline: fetch(s+1) overlaps WMMA(s); 1 barrier/step ----
    fetch(0);
    stash(0);
    __syncthreads();

    for (int s = 0; s < NSTEPS; ++s) {
        const int cur = s & 1;
        if (s + 1 < NSTEPS) fetch(s + 1);        // loads in flight under WMMA

        // A fragments (16x32 bf16): VGPR0..3 = K[half*8..+7], VGPR4..7 = K[16+half*8..+7]
        v16bf afrag[2];
#pragma unroll
        for (int mt = 0; mt < 2; ++mt) {
            const int row = wm * 32 + mt * 16 + l15;
            const v8bf lo = *(const v8bf*)&lds_x[cur][row][half * 8];
            const v8bf hi = *(const v8bf*)&lds_x[cur][row][16 + half * 8];
            afrag[mt] = __builtin_shufflevector(lo, hi,
                0, 1, 2, 3, 4, 5, 6, 7, 8, 9, 10, 11, 12, 13, 14, 15);
        }
        // B fragments (32x16 bf16): VGPR0..3 = K[half*8..+7], VGPR4..7 = K[16+half*8..+7]
#pragma unroll
        for (int g = 0; g < 4; ++g) {
            const int nl = g * BN + wn * 16 + l15;
            const v8bf lo = *(const v8bf*)&lds_w[cur][nl][half * 8];
            const v8bf hi = *(const v8bf*)&lds_w[cur][nl][16 + half * 8];
            const v16bf bfrag = __builtin_shufflevector(lo, hi,
                0, 1, 2, 3, 4, 5, 6, 7, 8, 9, 10, 11, 12, 13, 14, 15);
#pragma unroll
            for (int mt = 0; mt < 2; ++mt)
                acc[mt][g] = __builtin_amdgcn_wmma_f32_16x16x32_bf16(
                    false, afrag[mt], false, bfrag,
                    (short)0, acc[mt][g], false, false);
        }

        if (s + 1 < NSTEPS) stash(cur ^ 1);      // write the other buffer
        __syncthreads();
    }

    // ---- fused LSTM epilogue: wave-local a,i,f,o for the same (row, j) -----
    // C/D layout: VGPR v -> M = v + 8*half ; N = lane&15.
    const int j  = n0 + wn * 16 + l15;           // gate column (0..1023)
    const float ba = bias[0 * TRACKN + j];
    const float bi = bias[1 * TRACKN + j];
    const float bf = bias[2 * TRACKN + j];
    const float bo = bias[3 * TRACKN + j];

    // thread-constant bases; (mt,v) parts fold into instruction immediates
    const int r0 = m0 + wm * 32 + half * 8;
    const float* __restrict__ cp_base = c_prev + (size_t)r0 * TRACKN + j;
    float*       __restrict__ h_base  = h_out  + (size_t)r0 * TRACKN + j;
    float*       __restrict__ c_base  = c_out  + (size_t)r0 * TRACKN + j;

#pragma unroll
    for (int mt = 0; mt < 2; ++mt) {
#pragma unroll
        for (int v = 0; v < 8; ++v) {
            const int dr = mt * 16 + v;          // compile-time per unrolled iter
            const float a  = acc[mt][0][v] + ba;
            const float ig = acc[mt][1][v] + bi;
            const float fg = acc[mt][2][v] + bf;
            const float og = acc[mt][3][v] + bo;
            const float cp = cp_base[(size_t)dr * TRACKN];
            const float cc = fast_tanh(a) * fast_sigmoid(ig) + fast_sigmoid(fg) * cp;
            const float hh = fast_sigmoid(og) * fast_tanh(cc);
            h_base[(size_t)dr * TRACKN] = hh;
            c_base[(size_t)dr * TRACKN] = cc;
        }
    }
}

extern "C" void kernel_launch(void* const* d_in, const int* in_sizes, int n_in,
                              void* d_out, int out_size, void* d_ws, size_t ws_size,
                              hipStream_t stream) {
    (void)in_sizes; (void)n_in; (void)out_size; (void)d_ws; (void)ws_size;
    // setup_inputs order:
    // 0 top_buf_h, 1 top_stack1_h, 2 top_stack2_h, 3 h_prev, 4 c_prev,
    // 5 W_buf, 6 W_stack1, 7 W_stack2, 8 W_lateral, 9 b_lateral
    const float* x0   = (const float*)d_in[0];
    const float* x1   = (const float*)d_in[1];
    const float* x2   = (const float*)d_in[2];
    const float* x3   = (const float*)d_in[3];
    const float* cprv = (const float*)d_in[4];
    const float* w0   = (const float*)d_in[5];
    const float* w1   = (const float*)d_in[6];
    const float* w2   = (const float*)d_in[7];
    const float* w3   = (const float*)d_in[8];
    const float* bias = (const float*)d_in[9];

    float* h_out = (float*)d_out;                    // [4096,1024]
    float* c_out = h_out + (size_t)BATCH * TRACKN;   // [4096,1024]

    dim3 grid(TRACKN / BN, BATCH / BM);   // (32, 32); N fastest for L2 reuse of X
    spinn_tracker_wmma<<<grid, 256, 0, stream>>>(x0, x1, x2, x3,
                                                 w0, w1, w2, w3,
                                                 bias, cprv, h_out, c_out);
}
